// EdgeUpdateBlock_17394617549526
// MI455X (gfx1250) — matrix operand order repaired and checked by reading, not compiled, not run on hardware
//
#include <hip/hip_runtime.h>
#include <hip/hip_bf16.h>
#include <math.h>

typedef __bf16 bf16_t;
typedef __attribute__((ext_vector_type(16))) __bf16 v16bf;
typedef __attribute__((ext_vector_type(8)))  float  v8f;

#define WAVES   8
#define NBLK    512
#define A_W     196   // padded LDS row strides (multiples of 4 elems -> 8B-aligned rows)
#define B_W     132
#define W_W     196
#define NFRAG   168
#define FRAG_BYTES (NFRAG * 1024)
#define NG      16

#define wsync() __asm__ volatile("" ::: "memory")

// fragment-set base offsets (1 frag = 32x16 (k x n) bf16 tile in native B layout, 1KB)
#define FB_PRES  0
#define FB_PREV  8
#define FB_WSS   10
#define FB_WVV   46
#define FB_WSV   64
#define FB_WVS   76
#define FB_FC1   82
#define FB_FC2   98
#define FB_FC3   106
#define FB_POSTS 118
#define FB_POSTV 126
#define FB_SCS   128   // 4 sets of 8 (per one-hot column b)
#define FB_SCV   160   // 4 sets of 2

// ---------------- WMMA helpers ----------------

// A-fragment (16x32 bf16) from LDS row-major buffer.
// lanes 0-15: rows 0-15, K 0..15 ; lanes 16-31: same rows, K 16..31
__device__ static inline v16bf ldA(const bf16_t* base, int stride, int kbase, int lane) {
  const int row = lane & 15, half = lane >> 4;
  const bf16_t* p = base + row * stride + kbase + (half << 4);
  v16bf a;
#pragma unroll
  for (int e = 0; e < 16; ++e)
    a[e] = p[e];
  return a;
}

// D = A*B + C over KT k-tiles, NT n-tiles; B pre-packed fragment-native in LDS
template<int KT, int NT>
__device__ static inline void gemm(const bf16_t* aBase, int stride,
                                   const bf16_t* wgt, int fragBase,
                                   v8f* acc, int lane) {
#pragma unroll
  for (int kt = 0; kt < KT; ++kt) {
    v16bf a = ldA(aBase, stride, kt * 32, lane);
#pragma unroll
    for (int nt = 0; nt < NT; ++nt) {
      v16bf b = *(const v16bf*)(wgt + ((size_t)(fragBase + kt * NT + nt) << 9) + (lane << 4));
      acc[nt] = __builtin_amdgcn_wmma_f32_16x16x32_bf16(false, a, false, b,
                                                        (short)0, acc[nt], false, false);
    }
  }
}

// store C tiles (f32 accum) to LDS as bf16; C layout: col=lane&15 (+16*nt), row=r+8*(lane>>4)
template<int NT>
__device__ static inline void stC(bf16_t* dst, int stride, int colb,
                                  const v8f* acc, int lane, float scale) {
  const int col = colb + (lane & 15), half = lane >> 4;
#pragma unroll
  for (int nt = 0; nt < NT; ++nt)
#pragma unroll
    for (int r = 0; r < 8; ++r)
      dst[(r + (half << 3)) * stride + col + nt * 16] = (bf16_t)(acc[nt][r] * scale);
}

template<int NT>
__device__ static inline void addBias(v8f* acc, const float* bias, int lane) {
  const int col = lane & 15;
#pragma unroll
  for (int nt = 0; nt < NT; ++nt) {
    float b = bias[col + nt * 16];
#pragma unroll
    for (int r = 0; r < 8; ++r) acc[nt][r] += b;
  }
}

template<int NT>
__device__ static inline void addBiasSilu(v8f* acc, const float* bias, int lane) {
  const int col = lane & 15;
#pragma unroll
  for (int nt = 0; nt < NT; ++nt) {
    float b = bias[col + nt * 16];
#pragma unroll
    for (int r = 0; r < 8; ++r) {
      float x = acc[nt][r] + b;
      acc[nt][r] = x / (1.f + __expf(-x));
    }
  }
}

// acc += rowScalar[row][b] * t   (one-hot weighted skip-connection combine)
template<int NT>
__device__ static inline void axpyOH(v8f* acc, const v8f* t, const float (*oh)[4], int b, int lane) {
  const int half = lane >> 4;
#pragma unroll
  for (int nt = 0; nt < NT; ++nt)
#pragma unroll
    for (int r = 0; r < 8; ++r)
      acc[nt][r] += oh[r + (half << 3)][b] * t[nt][r];
}

// cooperative 16-row tile loads: global f32 -> LDS bf16 (scalar, for strided channels)
template<int C>
__device__ static inline void ldRowsE(bf16_t* dst, int dstride, const float* src, int gstride,
                                      int colOff, int estride, int e0, int lane) {
  for (int t = lane; t < 16 * C; t += 32) {
    int r = t / C, cc = t % C;
    dst[r * dstride + cc] = (bf16_t)(src[(size_t)(e0 + r) * gstride + colOff + cc * estride]);
  }
}
template<int C>
__device__ static inline void ldRowsG(bf16_t* dst, int dstride, const float* src, int gstride,
                                      int colOff, int estride, const int* rowIdx, int lane) {
  for (int t = lane; t < 16 * C; t += 32) {
    int r = t / C, cc = t % C;
    dst[r * dstride + cc] = (bf16_t)(src[(size_t)rowIdx[r] * gstride + colOff + cc * estride]);
  }
}

// vectorized variants for contiguous columns: float4 global loads
template<int C>
__device__ static inline void ldRowsE4(bf16_t* dst, int dstride, const float* src, int gstride,
                                       int colOff, int e0, int lane) {
  constexpr int CQ = C / 4;
  for (int t = lane; t < 16 * CQ; t += 32) {
    int r = t / CQ, cc = (t % CQ) * 4;
    const float4 v = *(const float4*)(src + (size_t)(e0 + r) * gstride + colOff + cc);
    bf16_t* d = dst + r * dstride + cc;
    d[0] = (bf16_t)v.x; d[1] = (bf16_t)v.y; d[2] = (bf16_t)v.z; d[3] = (bf16_t)v.w;
  }
}
template<int C>
__device__ static inline void ldRowsG4(bf16_t* dst, int dstride, const float* src, int gstride,
                                       int colOff, const int* rowIdx, int lane) {
  constexpr int CQ = C / 4;
  for (int t = lane; t < 16 * CQ; t += 32) {
    int r = t / CQ, cc = (t % CQ) * 4;
    const float4 v = *(const float4*)(src + (size_t)rowIdx[r] * gstride + colOff + cc);
    bf16_t* d = dst + r * dstride + cc;
    d[0] = (bf16_t)v.x; d[1] = (bf16_t)v.y; d[2] = (bf16_t)v.z; d[3] = (bf16_t)v.w;
  }
}

// ---------------- weight fragment packer ----------------

__constant__ int   g_base[19]   = {0,8,10,46,64,76,82,98,106,118,126,128,136,144,152,160,162,164,166};
__constant__ int   g_NT[19]     = {4,2,6,6,2,2,4,4,6,4,2, 4,4,4,4, 2,2,2,2};
__constant__ int   g_src[19]    = {0,1,2,3,4,5,6,7,8,9,10, 11,11,11,11, 12,12,12,12};
__constant__ int   g_stride[19] = {64,32,96,96,32,32,64,64,96,64,32, 256,256,256,256, 128,128,128,128};
__constant__ int   g_off[19]    = {0,0,0,0,0,0,0,0,0,0,0, 0,64,128,192, 0,32,64,96};
__constant__ float g_scale[19]  = {
  0.125f,               // pre_s   1/sqrt(64)
  0.1767766953f,        // pre_v   1/sqrt(32)
  0.0510310363f,        // w_ss    1/sqrt(192*2)
  0.0721687836f,        // w_vv    1/sqrt(96*2)
  0.0510310363f,        // w_sv    1/sqrt(192*2)
  0.0721687836f,        // w_vs    1/sqrt(96*2)
  0.0883883476f,        // fc1     1/sqrt(128)
  0.125f, 0.125f,       // fc2, fc3
  0.125f,               // post_s
  0.1767766953f,        // post_v
  0.0625f, 0.0625f, 0.0625f, 0.0625f,                     // sc_s 1/16
  0.0883883476f, 0.0883883476f, 0.0883883476f, 0.0883883476f  // sc_v
};

__global__ void kPack(const float* Wpre_s, const float* Wpre_v, const float* w_ss,
                      const float* w_vv, const float* w_sv, const float* w_vs,
                      const float* Wfc1, const float* Wfc2, const float* Wfc3,
                      const float* Wpost_s, const float* Wpost_v,
                      const float* Wsc_s, const float* Wsc_v, bf16_t* frags) {
  int t = blockIdx.x * blockDim.x + threadIdx.x;
  int frag = t >> 5, lane = t & 31;
  if (frag >= NFRAG) return;
  int s = 18;
  while (frag < g_base[s]) --s;
  int rel = frag - g_base[s];
  int NT = g_NT[s];
  int kt = rel / NT, nt = rel % NT;
  const float* src;
  switch (g_src[s]) {
    case 0:  src = Wpre_s;  break;  case 1:  src = Wpre_v;  break;
    case 2:  src = w_ss;    break;  case 3:  src = w_vv;    break;
    case 4:  src = w_sv;    break;  case 5:  src = w_vs;    break;
    case 6:  src = Wfc1;    break;  case 7:  src = Wfc2;    break;
    case 8:  src = Wfc3;    break;  case 9:  src = Wpost_s; break;
    case 10: src = Wpost_v; break;  case 11: src = Wsc_s;   break;
    default: src = Wsc_v;   break;
  }
  int col = nt * 16 + (lane & 15);
  int kb  = kt * 32 + (lane >> 4) * 16;     // B layout: lanes 0-15 K 0..15, lanes 16-31 K 16..31
  float sc = g_scale[s];
  bf16_t* dst = frags + ((size_t)frag << 9) + (lane << 4);
#pragma unroll
  for (int e = 0; e < 16; ++e)
    dst[e] = (bf16_t)(src[(size_t)g_off[s] + (size_t)(kb + e) * g_stride[s] + col] * sc);
}

// ---------------- main fused edge kernel ----------------

__global__ __launch_bounds__(WAVES * 32, 1)
void kMain(const float* __restrict__ node_fea, const float* __restrict__ one_hot,
           const float* __restrict__ esh, const float* __restrict__ efea,
           const float* __restrict__ ele, const int* __restrict__ eidx,
           const int* __restrict__ batch,
           const float* __restrict__ bpre_s, const float* __restrict__ bfc1,
           const float* __restrict__ bfc2, const float* __restrict__ bfc3,
           const float* __restrict__ bpost_s,
           const bf16_t* __restrict__ frags, float* __restrict__ q,
           float* __restrict__ gstat, int E) {
  __shared__ __attribute__((aligned(32))) bf16_t sWgt[NFRAG * 512];   // 168 KB packed weights
  __shared__ __attribute__((aligned(32))) bf16_t sAb[WAVES][16 * A_W];
  __shared__ __attribute__((aligned(32))) bf16_t sBb[WAVES][16 * B_W];
  __shared__ __attribute__((aligned(32))) bf16_t sWb[WAVES][16 * W_W];
  __shared__ float sSH[WAVES][16][4];
  __shared__ float sOH[WAVES][16][4];
  __shared__ int   sIJ[WAVES][2][16];
  __shared__ int   sG[WAVES][16];
  __shared__ float sStat[NG * 8];

  const int tid = threadIdx.x, w = tid >> 5, lane = tid & 31;
  const int half = lane >> 4, col15 = lane & 15;

  if (tid < NG * 8) sStat[tid] = 0.f;
  {  // stage packed weights to LDS
    const int4* src = (const int4*)frags;
    int4* dst = (int4*)sWgt;
    for (int t = tid; t < FRAG_BYTES / 16; t += WAVES * 32) dst[t] = src[t];
  }
  __syncthreads();

  bf16_t* Ab = &sAb[w][0];
  bf16_t* Bb = &sBb[w][0];
  bf16_t* Wb = &sWb[w][0];
  const bf16_t* WG = sWgt;
  const int nTiles = E / 16;
  const int tstep = gridDim.x * WAVES;

  for (int tile = blockIdx.x * WAVES + w; tile < nTiles; tile += tstep) {
    const int e0 = tile * 16;

    // prefetch next tile's streaming rows (global_prefetch_b8)
    if (tile + tstep < nTiles) {
      const size_t ne = (size_t)(tile + tstep) * 16 + col15;
      __builtin_prefetch(&efea[ne * 160], 0, 1);
      __builtin_prefetch(&efea[ne * 160 + 96], 0, 1);
      __builtin_prefetch(&ele[ne * 128], 0, 1);
      __builtin_prefetch(&ele[ne * 128 + 64], 0, 1);
    }

    // ---- S0: per-edge indices / sh / one-hot ----
    if (lane < 16) {
      int e = e0 + lane;
      int ii = eidx[e], jj = eidx[E + e];
      sIJ[w][0][lane] = ii; sIJ[w][1][lane] = jj;
      sG[w][lane] = batch[ii];
#pragma unroll
      for (int c = 0; c < 4; ++c) {
        sSH[w][lane][c] = esh[(size_t)e * 4 + c];
        sOH[w][lane][c] = one_hot[(size_t)e * 4 + c];
      }
    }
    wsync();

    // ---- S1: radial MLP (fc1->fc2->fc3), result w -> Wb[0..95] ----
    ldRowsE4<128>(Ab, A_W, ele, 128, 0, e0, lane); wsync();
    {
      v8f h[4] = {};
      gemm<4, 4>(Ab, A_W, WG, FB_FC1, h, lane);
      addBiasSilu<4>(h, bfc1, lane);
      stC<4>(Ab, A_W, 128, h, lane, 1.f); wsync();
      v8f h2[4] = {};
      gemm<2, 4>(Ab + 128, A_W, WG, FB_FC2, h2, lane);
      addBiasSilu<4>(h2, bfc2, lane);
      stC<4>(Ab, A_W, 0, h2, lane, 1.f); wsync();
      v8f wv[6] = {};
      gemm<2, 6>(Ab, A_W, WG, FB_FC3, wv, lane);
      addBias<6>(wv, bfc3, lane);
      stC<6>(Wb, W_W, 0, wv, lane, 1.f);
    }

    // ---- S2: es -> Bb[0..63]; pre_s -> s1[:,128:192] ----
    ldRowsE4<64>(Bb, B_W, efea, 160, 0, e0, lane); wsync();
    {
      v8f ps[4] = {};
      gemm<2, 4>(Bb, B_W, WG, FB_PRES, ps, lane);
      addBias<4>(ps, bpre_s, lane);
      stC<4>(Ab, A_W, 128, ps, lane, 1.f);
    }

    // ---- S3: gather ns[i], ns[j] -> s1[:,0:128] ----
    ldRowsG4<64>(Ab,      A_W, node_fea, 160, 0, sIJ[w][0], lane);
    ldRowsG4<64>(Ab + 64, A_W, node_fea, 160, 0, sIJ[w][1], lane);
    wsync();

    // ---- S4: sv = s1 @ w_sv  (registers across the c loop) ----
    v8f sv2[2] = {};
    gemm<6, 2>(Ab, A_W, WG, FB_WSV, sv2, lane);

    // ---- per-component c loop: v1, pre_v, dot, w_vs, zv_pre ----
    v8f dot6[6] = {};
#pragma unroll
    for (int c = 0; c < 3; ++c) {
      ldRowsG<32>(Bb,      B_W, node_fea, 160, 64 + c, 3, sIJ[w][0], lane);
      ldRowsG<32>(Bb + 32, B_W, node_fea, 160, 64 + c, 3, sIJ[w][1], lane);
      ldRowsE<32>(Bb + 96, B_W, efea,     160, 64 + c, 3, e0, lane);
      wsync();
      {
        v8f pvc[2] = {};
        gemm<1, 2>(Bb + 96, B_W, WG, FB_PREV, pvc, lane);
        stC<2>(Bb, B_W, 64, pvc, lane, 1.f);   // v1[:,64:96] = pv_c
      }
      wsync();
      // dot[u] += v1[u,c] * sh_v[c]
#pragma unroll
      for (int nt = 0; nt < 6; ++nt)
#pragma unroll
        for (int r = 0; r < 8; ++r) {
          int row = r + (half << 3);
          dot6[nt][r] += (float)Bb[row * B_W + nt * 16 + col15] * sSH[w][row][1 + c];
        }
      v8f vs2[2] = {};
      gemm<3, 2>(Bb, B_W, WG, FB_WVS, vs2, lane);
      // zv_pre = sv*sh_v[c] + (v1@w_vs)*sh_s   (1/sqrt2 folded into weights)
#pragma unroll
      for (int nt = 0; nt < 2; ++nt)
#pragma unroll
        for (int r = 0; r < 8; ++r) {
          int row = r + (half << 3);
          float shv = sSH[w][row][1 + c], shs = sSH[w][row][0];
          Wb[row * W_W + 96 + 32 * c + nt * 16 + col15] =
              (bf16_t)(sv2[nt][r] * shv + vs2[nt][r] * shs);
        }
    }

    // ---- S6: z_s = sh_s*(s1@w_ss) + dot@w_vv ; gates ----
    stC<6>(Bb, B_W, 0, dot6, lane, 0.57735026919f);  // dot/sqrt(3) -> Bb[0..95]
    wsync();
    v8f zs[6] = {};
    gemm<6, 6>(Ab, A_W, WG, FB_WSS, zs, lane);
#pragma unroll
    for (int nt = 0; nt < 6; ++nt)
#pragma unroll
      for (int r = 0; r < 8; ++r)
        zs[nt][r] *= sSH[w][r + (half << 3)][0];
    gemm<3, 6>(Bb, B_W, WG, FB_WVV, zs, lane);

    // c_s = silu(z_s[:, :64]) * w[:, :64]  -> Bb[0..63]
#pragma unroll
    for (int nt = 0; nt < 4; ++nt)
#pragma unroll
      for (int r = 0; r < 8; ++r) {
        int row = r + (half << 3);
        float x = zs[nt][r];
        float g = x / (1.f + __expf(-x));
        float ww = (float)Wb[row * W_W + nt * 16 + col15];
        Bb[row * B_W + nt * 16 + col15] = (bf16_t)(g * ww);
      }
    float sig[2][8];
#pragma unroll
    for (int nt2 = 0; nt2 < 2; ++nt2)
#pragma unroll
      for (int r = 0; r < 8; ++r)
        sig[nt2][r] = 1.f / (1.f + __expf(-zs[4 + nt2][r]));

    // es re-stage for the deferred sc_s skip-connection GEMM
    ldRowsE4<64>(Bb + 64, B_W, efea, 160, 0, e0, lane);
    wsync();

    // ---- qs = c_s@Wpost_s + bpost_s + sc_s(es,oh) ; write + stats ----
    float ssum[8] = {}, ssq[8] = {};
    {
      v8f qs4[4] = {};
      gemm<2, 4>(Bb, B_W, WG, FB_POSTS, qs4, lane);
#pragma unroll
      for (int b = 0; b < 4; ++b) {
        v8f tb[4] = {};
        gemm<2, 4>(Bb + 64, B_W, WG, FB_SCS + 8 * b, tb, lane);
        axpyOH<4>(qs4, tb, sOH[w], b, lane);
      }
      addBias<4>(qs4, bpost_s, lane);
#pragma unroll
      for (int nt = 0; nt < 4; ++nt)
#pragma unroll
        for (int r = 0; r < 8; ++r) {
          int row = r + (half << 3);
          float v = qs4[nt][r];
          q[(size_t)(e0 + row) * 160 + nt * 16 + col15] = v;
          ssum[r] += v; ssq[r] += v * v;
        }
    }

    // ---- qv_c = (zv_pre*sigmoid*w)@Wpost_v + sc_v(ev,oh) ; write + stats ----
    float vsum[3][8] = {}, vsq[8] = {};
#pragma unroll
    for (int c = 0; c < 3; ++c) {
      // c_v -> Bb[96..127]
#pragma unroll
      for (int nt2 = 0; nt2 < 2; ++nt2)
#pragma unroll
        for (int r = 0; r < 8; ++r) {
          int row = r + (half << 3);
          int u = nt2 * 16 + col15;
          float zv = (float)Wb[row * W_W + 96 + 32 * c + u];
          float ww = (float)Wb[row * W_W + 64 + u];
          Bb[row * B_W + 96 + u] = (bf16_t)(zv * sig[nt2][r] * ww);
        }
      // ev_c -> Bb[0..31] for deferred sc_v
      ldRowsE<32>(Bb, B_W, efea, 160, 64 + c, 3, e0, lane);
      wsync();
      v8f qv2[2] = {};
      gemm<1, 2>(Bb + 96, B_W, WG, FB_POSTV, qv2, lane);
#pragma unroll
      for (int b = 0; b < 4; ++b) {
        v8f tb[2] = {};
        gemm<1, 2>(Bb, B_W, WG, FB_SCV + 2 * b, tb, lane);
        axpyOH<2>(qv2, tb, sOH[w], b, lane);
      }
#pragma unroll
      for (int nt = 0; nt < 2; ++nt)
#pragma unroll
        for (int r = 0; r < 8; ++r) {
          int row = r + (half << 3);
          int u = nt * 16 + col15;
          float v = qv2[nt][r];
          q[(size_t)(e0 + row) * 160 + 64 + u * 3 + c] = v;
          vsum[c][r] += v; vsq[r] += v * v;
        }
    }

    // ---- per-row cross-lane reduce, accumulate graph stats in LDS ----
#pragma unroll
    for (int r = 0; r < 8; ++r) {
      float a0 = ssum[r], a1 = ssq[r];
      float a2 = vsum[0][r], a3 = vsum[1][r], a4 = vsum[2][r], a5 = vsq[r];
#pragma unroll
      for (int m = 1; m < 16; m <<= 1) {
        a0 += __shfl_xor(a0, m, 16); a1 += __shfl_xor(a1, m, 16);
        a2 += __shfl_xor(a2, m, 16); a3 += __shfl_xor(a3, m, 16);
        a4 += __shfl_xor(a4, m, 16); a5 += __shfl_xor(a5, m, 16);
      }
      if (col15 == 0) {
        int row = r + (half << 3);
        float* st = &sStat[sG[w][row] * 8];
        atomicAdd(st + 0, a0); atomicAdd(st + 1, a1);
        atomicAdd(st + 2, a2); atomicAdd(st + 3, a3); atomicAdd(st + 4, a4);
        atomicAdd(st + 5, a5); atomicAdd(st + 6, 1.0f);
      }
    }
  }

  __syncthreads();
  if (tid < NG * 8) atomicAdd(&gstat[tid], sStat[tid]);
}

// ---------------- small epilogue kernels ----------------

__global__ void kZero(float* stat) {
  if (threadIdx.x < 128) stat[threadIdx.x] = 0.f;
}

__global__ void kFinal(const float* stat, float* nrm) {
  int g = threadIdx.x;
  if (g >= NG) return;
  const float* s = stat + g * 8;
  float cnt = fmaxf(s[6], 1.f);
  float ms = s[0] / (64.f * cnt);
  float vs = s[1] / (64.f * cnt) - ms * ms;
  float m0 = s[2] / (32.f * cnt), m1 = s[3] / (32.f * cnt), m2 = s[4] / (32.f * cnt);
  float vv = s[5] / (96.f * cnt) - (m0 * m0 + m1 * m1 + m2 * m2) / 3.f;
  float* n = nrm + g * 8;
  n[0] = ms; n[1] = 1.f / (sqrtf(fmaxf(vs, 0.f)) + 1e-5f);
  n[2] = m0; n[3] = m1; n[4] = m2;
  n[5] = 1.f / (sqrtf(fmaxf(vv, 0.f)) + 1e-5f);
}

__global__ void kNorm(const float* __restrict__ efea, const int* __restrict__ eidx,
                      const int* __restrict__ batch, const float* __restrict__ q,
                      const float* __restrict__ nrm, const float* __restrict__ gamma_s,
                      const float* __restrict__ beta_s, const float* __restrict__ gamma_v,
                      float* __restrict__ out, int E) {
  int idx = blockIdx.x * blockDim.x + threadIdx.x;
  if (idx >= E * 160) return;
  int e = idx / 160, f = idx % 160;
  const float* p = nrm + batch[eidx[e]] * 8;
  float qq = q[idx], r;
  if (f < 64) {
    r = (qq - p[0]) * p[1] * gamma_s[f] + beta_s[f];
  } else {
    int u = (f - 64) / 3, c = (f - 64) % 3;
    r = (qq - p[2 + c]) * p[5] * gamma_v[u];
  }
  out[idx] = efea[idx] + r;
}

// ---------------- host entry ----------------

extern "C" void kernel_launch(void* const* d_in, const int* in_sizes, int n_in,
                              void* d_out, int out_size, void* d_ws, size_t ws_size,
                              hipStream_t stream) {
  const float* node_fea = (const float*)d_in[0];
  const float* one_hot  = (const float*)d_in[1];
  const float* esh      = (const float*)d_in[2];
  const float* efea     = (const float*)d_in[3];
  const float* ele      = (const float*)d_in[4];
  const int*   eidx     = (const int*)d_in[5];
  const int*   batch    = (const int*)d_in[6];
  const float* Wpre_s = (const float*)d_in[7];
  const float* bpre_s = (const float*)d_in[8];
  const float* Wpre_v = (const float*)d_in[9];
  const float* w_ss   = (const float*)d_in[10];
  const float* w_vv   = (const float*)d_in[11];
  const float* w_sv   = (const float*)d_in[12];
  const float* w_vs   = (const float*)d_in[13];
  const float* Wfc1 = (const float*)d_in[14];
  const float* bfc1 = (const float*)d_in[15];
  const float* Wfc2 = (const float*)d_in[16];
  const float* bfc2 = (const float*)d_in[17];
  const float* Wfc3 = (const float*)d_in[18];
  const float* bfc3 = (const float*)d_in[19];
  const float* Wsc_s   = (const float*)d_in[20];
  const float* Wsc_v   = (const float*)d_in[21];
  const float* Wpost_s = (const float*)d_in[22];
  const float* bpost_s = (const float*)d_in[23];
  const float* Wpost_v = (const float*)d_in[24];
  const float* gamma_s = (const float*)d_in[25];
  const float* beta_s  = (const float*)d_in[26];
  const float* gamma_v = (const float*)d_in[27];

  const int E = in_sizes[3] / 160;

  char* ws = (char*)d_ws;
  float*  q     = (float*)ws;
  bf16_t* frags = (bf16_t*)(ws + (size_t)E * 160 * 4);
  float*  stat  = (float*)(ws + (size_t)E * 160 * 4 + FRAG_BYTES);
  float*  nrm   = stat + 128;

  kZero<<<1, 128, 0, stream>>>(stat);
  kPack<<<(NFRAG * 32 + 255) / 256, 256, 0, stream>>>(
      Wpre_s, Wpre_v, w_ss, w_vv, w_sv, w_vs, Wfc1, Wfc2, Wfc3,
      Wpost_s, Wpost_v, Wsc_s, Wsc_v, frags);
  kMain<<<NBLK, WAVES * 32, 0, stream>>>(
      node_fea, one_hot, esh, efea, ele, eidx, batch,
      bpre_s, bfc1, bfc2, bfc3, bpost_s, frags, q, stat, E);
  kFinal<<<1, 32, 0, stream>>>(stat, nrm);
  long tot = (long)E * 160;
  kNorm<<<(int)((tot + 255) / 256), 256, 0, stream>>>(
      efea, eidx, batch, q, nrm, gamma_s, beta_s, gamma_v, (float*)d_out, E);
}
